// CrossModalRefinementTransformer_86792699118198
// MI455X (gfx1250) — compile-verified, matrix operand order checked by reading, not run on hardware
//
#include <hip/hip_runtime.h>
#include <hip/hip_bf16.h>

typedef __attribute__((ext_vector_type(16))) _Float16 v16h;
typedef __attribute__((ext_vector_type(8)))  float    v8f;
typedef int v4i __attribute__((vector_size(16)));

#define TM 128
#define TN 128
#define TK 64
#define LDS_STRIDE 80   // 64 + 16 pad halves -> 160B row stride (32B aligned)

// ---- CDNA5 async global->LDS path (guarded; falls back to reg+ds_store) ----
#if defined(__AMDGCN__) && __has_builtin(__builtin_amdgcn_global_load_async_to_lds_b128) && \
    __has_builtin(__builtin_amdgcn_s_wait_asynccnt)
#define USE_ASYNC_LDS 1
#else
#define USE_ASYNC_LDS 0
#endif

#if USE_ASYNC_LDS
typedef __attribute__((address_space(1))) v4i GASv4i;
typedef __attribute__((address_space(3))) v4i LASv4i;
#endif

// ---------------- tile loaders ----------------
// 128 rows x 64 halves, row-major along K. 1024 x 16B, 4 per thread.
__device__ __forceinline__ void load_tile_nt(const _Float16* __restrict__ G,
                                             int base_row, int ld, int k0,
                                             _Float16* __restrict__ lds, int tid) {
#pragma unroll
    for (int i = 0; i < 4; ++i) {
        int linear = i * 256 + tid;
        int row = linear >> 3;
        int seg = linear & 7;
        const _Float16* g = G + (size_t)(base_row + row) * ld + k0 + seg * 8;
        _Float16* l = lds + row * LDS_STRIDE + seg * 8;
#if USE_ASYNC_LDS
        // ASYNCcnt-tracked DMA: 16B per lane straight into LDS, no VGPR staging.
        __builtin_amdgcn_global_load_async_to_lds_b128((GASv4i*)g, (LASv4i*)l, 0, 0);
#else
        *(uint4*)l = *(const uint4*)g;
#endif
    }
}

__device__ __forceinline__ void load_tile_tr(const _Float16* __restrict__ G,
                                             int n0, int N, int k0,
                                             _Float16* __restrict__ lds, int tid) {
    // B is KxN row-major; produce LDS tile [n][k] (transpose on LDS store).
#pragma unroll
    for (int i = 0; i < 4; ++i) {
        int linear = i * 256 + tid;       // 0..1023
        int k = linear & 63;              // 64 K rows
        int seg = linear >> 6;            // 16 segments of 8 cols
        union { uint4 u; _Float16 h[8]; } v;
        v.u = *(const uint4*)(G + (size_t)(k0 + k) * N + n0 + seg * 8);
#pragma unroll
        for (int j = 0; j < 8; ++j)
            lds[(seg * 8 + j) * LDS_STRIDE + k] = v.h[j];
    }
}

// Wait for this wave's outstanding tile fills, then workgroup barrier.
__device__ __forceinline__ void tile_fence() {
#if USE_ASYNC_LDS
    __builtin_amdgcn_s_wait_asynccnt(0);
#endif
    __syncthreads();
}

// ---------------- WMMA GEMM: out[m,n] = sum_k A[m,k] * B'[n,k] ----------------
// TRANSB=false: Bm is N x K row-major (i.e. x @ W.T with W row-major).
// TRANSB=true : Bm is K x N row-major (plain NN product).
template <bool TRANSB>
__global__ __launch_bounds__(256) void gemm_wmma(
    const _Float16* __restrict__ A, const _Float16* __restrict__ Bm,
    float* __restrict__ outF, _Float16* __restrict__ outH,
    const float* __restrict__ bias, const float* __restrict__ resid,
    int M, int N, int K, int relu)
{
    __shared__ __align__(32) _Float16 As[2][TM * LDS_STRIDE];
    __shared__ __align__(32) _Float16 Bs[2][TN * LDS_STRIDE];

    const int tid  = threadIdx.x;
    const int lane = tid & 31;
    const int wid  = tid >> 5;
    const int wm   = wid & 3;   // 4 wave-rows  (32 rows each)
    const int wn   = wid >> 2;  // 2 wave-cols  (64 cols each)
    const int m0   = blockIdx.y * TM;
    const int n0   = blockIdx.x * TN;

    v8f acc[2][4];
#pragma unroll
    for (int mt = 0; mt < 2; ++mt)
#pragma unroll
        for (int nt = 0; nt < 4; ++nt)
            acc[mt][nt] = (v8f){0.f, 0.f, 0.f, 0.f, 0.f, 0.f, 0.f, 0.f};

    const int KT = K / TK;

    load_tile_nt(A, m0, K, 0, &As[0][0], tid);
    if (!TRANSB) load_tile_nt(Bm, n0, K, 0, &Bs[0][0], tid);
    else         load_tile_tr(Bm, n0, N, 0, &Bs[0][0], tid);
    tile_fence();

    for (int kt = 0; kt < KT; ++kt) {
        const int cur = kt & 1;
        if (kt + 1 < KT) {
            load_tile_nt(A, m0, K, (kt + 1) * TK, &As[cur ^ 1][0], tid);
            if (!TRANSB) load_tile_nt(Bm, n0, K, (kt + 1) * TK, &Bs[cur ^ 1][0], tid);
            else         load_tile_tr(Bm, n0, N, (kt + 1) * TK, &Bs[cur ^ 1][0], tid);
        }
        if (kt + 2 < KT) {
            // Pull the tile after next into GL2 ahead of the async engine.
            const int k2 = (kt + 2) * TK;
            __builtin_prefetch(A + (size_t)(m0 + (tid >> 1)) * K + k2 + (tid & 1) * 32, 0, 1);
            if (!TRANSB)
                __builtin_prefetch(Bm + (size_t)(n0 + (tid >> 1)) * K + k2 + (tid & 1) * 32, 0, 1);
            else
                __builtin_prefetch(Bm + (size_t)(k2 + (tid >> 2)) * N + n0 + (tid & 3) * 32, 0, 1);
        }
        const int r  = lane & 15;
        const int hh = lane >> 4;
#pragma unroll
        for (int ks = 0; ks < 2; ++ks) {
            v16h af[2], bf[4];
#pragma unroll
            for (int mt = 0; mt < 2; ++mt)
                af[mt] = *(const v16h*)(&As[cur][(wm * 32 + mt * 16 + r) * LDS_STRIDE + ks * 32 + hh * 16]);
#pragma unroll
            for (int nt = 0; nt < 4; ++nt)
                bf[nt] = *(const v16h*)(&Bs[cur][(wn * 64 + nt * 16 + r) * LDS_STRIDE + ks * 32 + hh * 16]);
#pragma unroll
            for (int mt = 0; mt < 2; ++mt)
#pragma unroll
                for (int nt = 0; nt < 4; ++nt)
                    acc[mt][nt] = __builtin_amdgcn_wmma_f32_16x16x32_f16(
                        false, af[mt], false, bf[nt], (short)0, acc[mt][nt], false, false);
        }
        tile_fence();
    }

    // Epilogue: C/D layout -> VGPR j, lanes 0-15: M=j, lanes 16-31: M=j+8; N=lane%16.
#pragma unroll
    for (int mt = 0; mt < 2; ++mt) {
#pragma unroll
        for (int nt = 0; nt < 4; ++nt) {
            const int col   = n0 + wn * 64 + nt * 16 + (lane & 15);
            const int rbase = m0 + wm * 32 + mt * 16 + (lane >> 4) * 8;
            const float bv = bias ? bias[col] : 0.f;
#pragma unroll
            for (int j = 0; j < 8; ++j) {
                const int row = rbase + j;
                float v = acc[mt][nt][j] + bv;
                if (resid) v += resid[(size_t)row * N + col];
                if (relu)  v = fmaxf(v, 0.f);
                if (outF) outF[(size_t)row * N + col] = v;
                if (outH) outH[(size_t)row * N + col] = (_Float16)v;
            }
        }
    }
}

// ---------------- elementwise / row kernels ----------------
__global__ __launch_bounds__(256) void k_convert(const float* __restrict__ s,
                                                 _Float16* __restrict__ d, long long n) {
    for (long long i = (long long)blockIdx.x * 256 + threadIdx.x; i < n;
         i += (long long)gridDim.x * 256)
        d[i] = (_Float16)s[i];
}

__global__ __launch_bounds__(256) void k_add_pe(const float* __restrict__ s,
                                                const float* __restrict__ pe,
                                                float* __restrict__ o32,
                                                _Float16* __restrict__ o16, long long n) {
    for (long long i = (long long)blockIdx.x * 256 + threadIdx.x; i < n;
         i += (long long)gridDim.x * 256) {
        float v = s[i] + pe[i & 2047];
        if (o32) o32[i] = v;
        o16[i] = (_Float16)v;
    }
}

// row LayerNorm over D=2048; optional RoPE-style theta shift (row index = position)
__global__ __launch_bounds__(256) void k_layernorm(const float* __restrict__ in,
                                                   const float* __restrict__ g,
                                                   const float* __restrict__ b,
                                                   _Float16* __restrict__ out, int theta) {
    const int row = blockIdx.x, t = threadIdx.x;
    const float* v = in + (size_t)row * 2048;
    float loc[8];
    float s = 0.f, sq = 0.f;
#pragma unroll
    for (int i = 0; i < 8; ++i) { float x = v[t * 8 + i]; loc[i] = x; s += x; sq += x * x; }
    __shared__ float r1[256], r2[256];
    r1[t] = s; r2[t] = sq; __syncthreads();
    for (int off = 128; off; off >>= 1) {
        if (t < off) { r1[t] += r1[t + off]; r2[t] += r2[t + off]; }
        __syncthreads();
    }
    const float mu   = r1[0] * (1.f / 2048.f);
    const float var  = r2[0] * (1.f / 2048.f) - mu * mu;
    const float rstd = rsqrtf(var + 1e-5f);
    _Float16* o = out + (size_t)row * 2048;
    if (!theta) {
#pragma unroll
        for (int i = 0; i < 8; ++i) {
            int d = t * 8 + i;
            o[d] = (_Float16)((loc[i] - mu) * rstd * g[d] + b[d]);
        }
    } else {
#pragma unroll
        for (int p = 0; p < 4; ++p) {
            int d0 = t * 8 + p * 2;
            float n0 = (loc[p * 2]     - mu) * rstd * g[d0]     + b[d0];
            float n1 = (loc[p * 2 + 1] - mu) * rstd * g[d0 + 1] + b[d0 + 1];
            int j = d0 >> 1;
            float ang = (float)row * __powf(10000.f, -(float)j * (1.f / 1023.f));
            float sn, cs; __sincosf(ang, &sn, &cs);
            o[d0]     = (_Float16)(n0 * cs - n1 * sn);
            o[d0 + 1] = (_Float16)(n1 * cs + n0 * sn);
        }
    }
}

// row softmax over 8192 cols (decay bias cancels per-row; scale folded here).
// In-place f32 result (this is the returned A), plus f16 copy for the A@x_t GEMM.
__global__ __launch_bounds__(256) void k_softmax(float* __restrict__ S,
                                                 _Float16* __restrict__ P, float scale) {
    const int row = blockIdx.x, t = threadIdx.x;
    float* s = S + (size_t)row * 8192;
    float vals[32];
    float mx = -3.4e38f;
#pragma unroll
    for (int i = 0; i < 32; ++i) {
        float x = s[t + i * 256] * scale;
        vals[i] = x; mx = fmaxf(mx, x);
    }
    __shared__ float red[256];
    red[t] = mx; __syncthreads();
    for (int off = 128; off; off >>= 1) {
        if (t < off) red[t] = fmaxf(red[t], red[t + off]);
        __syncthreads();
    }
    mx = red[0]; __syncthreads();
    float sum = 0.f;
#pragma unroll
    for (int i = 0; i < 32; ++i) { vals[i] = __expf(vals[i] - mx); sum += vals[i]; }
    red[t] = sum; __syncthreads();
    for (int off = 128; off; off >>= 1) {
        if (t < off) red[t] += red[t + off];
        __syncthreads();
    }
    const float inv = 1.f / red[0];
#pragma unroll
    for (int i = 0; i < 32; ++i) {
        float p = vals[i] * inv;
        s[t + i * 256] = p;
        P[(size_t)row * 8192 + t + i * 256] = (_Float16)p;
    }
}

__global__ __launch_bounds__(256) void k_cat(const float* __restrict__ hnew,
                                             const _Float16* __restrict__ h16,
                                             _Float16* __restrict__ cat, long long n) {
    for (long long i = (long long)blockIdx.x * 256 + threadIdx.x; i < n;
         i += (long long)gridDim.x * 256) {
        long long row = i >> 11, d = i & 2047;
        cat[row * 4096 + d]        = (_Float16)hnew[i];
        cat[row * 4096 + 2048 + d] = h16[i];
    }
}

__global__ __launch_bounds__(256) void k_gate(const float* __restrict__ glog,
                                              const float* __restrict__ hnew,
                                              float* __restrict__ h32,
                                              _Float16* __restrict__ h16, long long n) {
    for (long long i = (long long)blockIdx.x * 256 + threadIdx.x; i < n;
         i += (long long)gridDim.x * 256) {
        float g = 1.f / (1.f + __expf(-glog[i]));
        float r = g * hnew[i] + (1.f - g) * h32[i];
        h32[i] = r;
        h16[i] = (_Float16)r;
    }
}

// h = LN(h2 + relu(t2)) with fln params; writes f32 + f16
__global__ __launch_bounds__(256) void k_add_relu_ln(const float* __restrict__ h2,
                                                     const float* __restrict__ t2,
                                                     const float* __restrict__ g,
                                                     const float* __restrict__ b,
                                                     float* __restrict__ outF,
                                                     _Float16* __restrict__ outH) {
    const int row = blockIdx.x, t = threadIdx.x;
    const float* a = h2 + (size_t)row * 2048;
    const float* r = t2 + (size_t)row * 2048;
    float loc[8];
    float s = 0.f, sq = 0.f;
#pragma unroll
    for (int i = 0; i < 8; ++i) {
        float x = a[t * 8 + i] + fmaxf(r[t * 8 + i], 0.f);
        loc[i] = x; s += x; sq += x * x;
    }
    __shared__ float r1[256], r2[256];
    r1[t] = s; r2[t] = sq; __syncthreads();
    for (int off = 128; off; off >>= 1) {
        if (t < off) { r1[t] += r1[t + off]; r2[t] += r2[t + off]; }
        __syncthreads();
    }
    const float mu   = r1[0] * (1.f / 2048.f);
    const float var  = r2[0] * (1.f / 2048.f) - mu * mu;
    const float rstd = rsqrtf(var + 1e-5f);
#pragma unroll
    for (int i = 0; i < 8; ++i) {
        int d = t * 8 + i;
        float v = (loc[i] - mu) * rstd * g[d] + b[d];
        outF[(size_t)row * 2048 + d] = v;
        outH[(size_t)row * 2048 + d] = (_Float16)v;
    }
}

// ---------------- orchestration ----------------
extern "C" void kernel_launch(void* const* d_in, const int* in_sizes, int n_in,
                              void* d_out, int out_size, void* d_ws, size_t ws_size,
                              hipStream_t stream) {
    const int Dd = 2048, Bz = 8192, Mz = 4096, D2 = 4096, D4 = 8192;
    const float scale = 0.022097086912079608f; // 1/sqrt(2048)

    const float* x      = (const float*)d_in[0];
    const float* y      = (const float*)d_in[1];
    const float* pe0    = (const float*)d_in[2];
    const float* attn_W = (const float*)d_in[3];
    const float* attn_b = (const float*)d_in[4];
    const float* ln_g   = (const float*)d_in[5];
    const float* ln_b   = (const float*)d_in[6];
    const float* gf_W   = (const float*)d_in[7];
    const float* gf_b   = (const float*)d_in[8];
    const float* res_W  = (const float*)d_in[9];
    const float* res_b  = (const float*)d_in[10];
    const float* ffn_W1 = (const float*)d_in[11];
    const float* ffn_b1 = (const float*)d_in[12];
    const float* ffn_W2 = (const float*)d_in[13];
    const float* ffn_b2 = (const float*)d_in[14];
    const float* fln_g  = (const float*)d_in[15];
    const float* fln_b  = (const float*)d_in[16];

    float* out_h = (float*)d_out;
    float* out_A = (float*)d_out + (size_t)Mz * Dd;   // doubles as the scores buffer

    char* wsb = (char*)d_ws;
    size_t off = 0;
    auto alloc = [&](size_t bytes) -> void* {
        void* p = wsb + off;
        off = (off + bytes + 255) & ~(size_t)255;
        return p;
    };

    _Float16* attnW16 = (_Float16*)alloc((size_t)2 * Dd * Dd * 2);
    _Float16* gfW16   = (_Float16*)alloc((size_t)2 * Dd * D2 * 2);
    _Float16* resW16  = (_Float16*)alloc((size_t)2 * Dd * Dd * 2);
    _Float16* W1_16   = (_Float16*)alloc((size_t)D4 * Dd * 2);
    _Float16* W2_16   = (_Float16*)alloc((size_t)Dd * D4 * 2);
    _Float16* xs16    = (_Float16*)alloc((size_t)Bz * Dd * 2);
    _Float16* xt16    = (_Float16*)alloc((size_t)Bz * Dd * 2);
    _Float16* yt16    = (_Float16*)alloc((size_t)Mz * Dd * 2);
    float*    raw32   = (float*)   alloc((size_t)Bz * Dd * 4);  // xt_raw / yt_raw / glog / t2
    float*    h32     = (float*)   alloc((size_t)Mz * Dd * 4);
    _Float16* h16     = (_Float16*)alloc((size_t)Mz * Dd * 2);
    _Float16* P16     = (_Float16*)alloc((size_t)Mz * Bz * 2);
    float*    hnew32  = (float*)   alloc((size_t)Mz * Dd * 4);
    _Float16* cat16   = (_Float16*)alloc((size_t)Mz * D2 * 2);
    _Float16* t1_16   = (_Float16*)alloc((size_t)Mz * D4 * 2);
    float*    h2_32   = (float*)   alloc((size_t)Mz * Dd * 4);
    _Float16* h2_16   = (_Float16*)alloc((size_t)Mz * Dd * 2);

    const dim3 blk(256);
    const int EW = 4096; // elementwise grid-stride blocks

    // weight conversion f32 -> f16
    k_convert<<<EW, blk, 0, stream>>>(attn_W, attnW16, (long long)2 * Dd * Dd);
    k_convert<<<EW, blk, 0, stream>>>(gf_W,   gfW16,   (long long)2 * Dd * D2);
    k_convert<<<EW, blk, 0, stream>>>(res_W,  resW16,  (long long)2 * Dd * Dd);
    k_convert<<<EW, blk, 0, stream>>>(ffn_W1, W1_16,   (long long)D4 * Dd);
    k_convert<<<EW, blk, 0, stream>>>(ffn_W2, W2_16,   (long long)Dd * D4);

    // xs = x + pe0 (f16 only); h = ys = y + pe0 (f32 + f16)
    k_add_pe<<<EW, blk, 0, stream>>>(x, pe0, nullptr, xs16, (long long)Bz * Dd);
    k_add_pe<<<EW, blk, 0, stream>>>(y, pe0, h32, h16, (long long)Mz * Dd);

    for (int i = 0; i < 2; ++i) {
        const _Float16* Wa = attnW16 + (size_t)i * Dd * Dd;

        // x_t = LN(xs @ Wa.T + ba)
        gemm_wmma<false><<<dim3(Dd / TN, Bz / TM), blk, 0, stream>>>(
            xs16, Wa, raw32, nullptr, attn_b + (size_t)i * Dd, nullptr, Bz, Dd, Dd, 0);
        k_layernorm<<<Bz, blk, 0, stream>>>(raw32, ln_g + (size_t)i * Dd,
                                            ln_b + (size_t)i * Dd, xt16, 0);

        // y_t = theta_shift(LN(h @ Wa.T + ba))
        gemm_wmma<false><<<dim3(Dd / TN, Mz / TM), blk, 0, stream>>>(
            h16, Wa, raw32, nullptr, attn_b + (size_t)i * Dd, nullptr, Mz, Dd, Dd, 0);
        k_layernorm<<<Mz, blk, 0, stream>>>(raw32, ln_g + (size_t)i * Dd,
                                            ln_b + (size_t)i * Dd, yt16, 1);

        // scores = y_t @ x_t.T  (NT); softmax (scale folded, decay bias cancels)
        gemm_wmma<false><<<dim3(Bz / TN, Mz / TM), blk, 0, stream>>>(
            yt16, xt16, out_A, nullptr, nullptr, nullptr, Mz, Bz, Dd, 0);
        k_softmax<<<Mz, blk, 0, stream>>>(out_A, P16, scale);

        // h_new = A @ x_t (NN: B is K x N)
        gemm_wmma<true><<<dim3(Dd / TN, Mz / TM), blk, 0, stream>>>(
            P16, xt16, hnew32, nullptr, nullptr, nullptr, Mz, Dd, Bz, 0);

        // gate = sigmoid([h_new, h] @ gf_W.T + gf_b); h = gate*h_new + (1-gate)*h
        k_cat<<<EW, blk, 0, stream>>>(hnew32, h16, cat16, (long long)Mz * Dd);
        gemm_wmma<false><<<dim3(Dd / TN, Mz / TM), blk, 0, stream>>>(
            cat16, gfW16 + (size_t)i * Dd * D2, raw32, nullptr,
            gf_b + (size_t)i * Dd, nullptr, Mz, Dd, D2, 0);
        k_gate<<<EW, blk, 0, stream>>>(raw32, hnew32, h32, h16, (long long)Mz * Dd);

        // h2 = h + relu(h @ W1.T + b1) @ W2.T + b2
        gemm_wmma<false><<<dim3(D4 / TN, Mz / TM), blk, 0, stream>>>(
            h16, W1_16, nullptr, t1_16, ffn_b1, nullptr, Mz, D4, Dd, 1);
        gemm_wmma<false><<<dim3(Dd / TN, Mz / TM), blk, 0, stream>>>(
            t1_16, W2_16, h2_32, h2_16, ffn_b2, h32, Mz, Dd, D4, 0);

        // t2 = h2 @ res_W.T + res_b ; h = LN(h2 + relu(t2))
        gemm_wmma<false><<<dim3(Dd / TN, Mz / TM), blk, 0, stream>>>(
            h2_16, resW16 + (size_t)i * Dd * Dd, raw32, nullptr,
            res_b + (size_t)i * Dd, nullptr, Mz, Dd, Dd, 0);
        k_add_relu_ln<<<Mz, blk, 0, stream>>>(h2_32, raw32, fln_g, fln_b,
                                              (i == 0) ? h32 : out_h, h16);
    }
}
